// GRU_66838281060501
// MI455X (gfx1250) — compile-verified, matrix operand order checked by reading, not run on hardware
//
#include <hip/hip_runtime.h>
#include <hip/hip_bf16.h>

#define BQ 16
#define HQ 512
#define TQ 512
#define G3 1536   // 3*H

typedef __attribute__((ext_vector_type(16))) _Float16 v16h;
typedef __attribute__((ext_vector_type(8)))  _Float16 v8h;
typedef __attribute__((ext_vector_type(8)))  float    v8f;

__device__ __forceinline__ v8f wmma_f16(v16h a, v16h b, v8f c) {
  // D = A(16x32 f16) * B(32x16 f16) + C(16x16 f32)
  return __builtin_amdgcn_wmma_f32_16x16x32_f16(false, a, false, b, (short)0, c, false, false);
}

// Assemble a 16-half fragment from a lane-resolved base pointer:
// 8 contiguous halves at q, 8 at q+16 (CDNA5 16-bit A/B K-split layout).
__device__ __forceinline__ v16h load16(const _Float16* q) {
  v8h lo = *(const v8h*)q;
  v8h hi = *(const v8h*)(q + 16);
  v16h a;
#pragma unroll
  for (int i = 0; i < 8; ++i) { a[i] = lo[i]; a[i + 8] = hi[i]; }
  return a;
}

// Fragment from row-major [rows, stride]: row = rowbase + (lane&15),
// K offset kc + (lane>=16 ? 8 : 0), plus the +16 second half inside load16.
__device__ __forceinline__ v16h load_frag(const _Float16* p, int rowbase, int stride,
                                          int kc, int lane) {
  return load16(p + (size_t)(rowbase + (lane & 15)) * stride + kc + ((lane & 16) ? 8 : 0));
}

__global__ void cvt_f32_f16(const float* __restrict__ src, _Float16* __restrict__ dst, int n) {
  int i = blockIdx.x * 256 + threadIdx.x;
  if (i < n) dst[i] = (_Float16)src[i];
}

// Y[M=8192, N=1536] = X[M,K](f16) * W[N,K]^T (f16) + bias (f32), f32 out.
// One wave computes a 16x64 strip (4 N-tiles sharing the A fragment).
__global__ __launch_bounds__(256) void gemm_xg(const _Float16* __restrict__ X,
                                               const _Float16* __restrict__ W,
                                               const float* __restrict__ bias,
                                               float* __restrict__ Y, int K) {
  int wid  = blockIdx.x * 8 + (threadIdx.x >> 5);
  int lane = threadIdx.x & 31;
  int tm = wid / 24;          // 512 M-tiles
  int n4 = wid % 24;          // 24 strips of 64 cols
  int nbase = n4 * 64;
  v8f zero = {0.f,0.f,0.f,0.f,0.f,0.f,0.f,0.f};
  v8f acc[4];
#pragma unroll
  for (int j = 0; j < 4; ++j) acc[j] = zero;
  for (int kc = 0; kc < K; kc += 32) {
    v16h a = load_frag(X, tm * 16, K, kc, lane);
#pragma unroll
    for (int j = 0; j < 4; ++j) {
      v16h b = load_frag(W, nbase + j * 16, K, kc, lane);
      acc[j] = wmma_f16(a, b, acc[j]);
    }
  }
  int nl = lane & 15, mb = (lane & 16) ? 8 : 0;
#pragma unroll
  for (int j = 0; j < 4; ++j) {
    int n0 = nbase + j * 16 + nl;
    float bv = bias[n0];
#pragma unroll
    for (int r = 0; r < 8; ++r) {
      int m = tm * 16 + r + mb;
      Y[(size_t)m * G3 + n0] = acc[j][r] + bv;
    }
  }
}

// Sequential GRU scan body, run by one 512-thread (16-wave) workgroup.
// h (B=16 x H=512) lives in LDS (f32 state + f16 WMMA mirror). Each wave owns
// 2 hidden-column tiles and computes r/z/n gate tiles (cols c, c+512, c+1024
// of gh = h@whh^T) with WMMA, then fuses the GRU cell in registers.
__device__ void gru_scan_core(float* hf, _Float16* hh,
                              const _Float16* __restrict__ whh,  // [1536,512] f16
                              const float* __restrict__ bhh,     // [1536]
                              const float* __restrict__ xg,      // [16,512,1536] f32
                              const float* __restrict__ h0,      // [16,512] f32
                              _Float16* __restrict__ outh,       // [16,512,out_stride] f16
                              int out_stride, int out_off,
                              float* __restrict__ outf,          // optional [16,512,512] f32
                              int reverse) {
  int tid = threadIdx.x;
  for (int i = tid; i < BQ * HQ; i += 512) {
    float v = h0[i];
    hf[i] = v;
    hh[i] = (_Float16)v;
  }
  __syncthreads();

  int lane = tid & 31, wave = tid >> 5;
  int nl = lane & 15, mb = (lane & 16) ? 8 : 0;
  int klo = (lane & 16) ? 8 : 0;

  // Timestep-invariant lane-resolved pointers / biases (hoisted out of t loop).
  const _Float16* aPtr = hh + nl * HQ + klo;        // A rows = batch (M=16)
  const _Float16* bPtr[2][3];
  float bias[2][3];
  int   col[2];
#pragma unroll
  for (int u = 0; u < 2; ++u) {
    int ht = wave * 2 + u;                          // hidden tile 0..31
    col[u] = ht * 16 + nl;
#pragma unroll
    for (int g = 0; g < 3; ++g) {
      bPtr[u][g] = whh + (size_t)(g * HQ + ht * 16 + nl) * HQ + klo;
      bias[u][g] = bhh[g * HQ + col[u]];
    }
  }
  v8f zero = {0.f,0.f,0.f,0.f,0.f,0.f,0.f,0.f};

  for (int tt = 0; tt < TQ; ++tt) {
    int t = reverse ? (TQ - 1 - tt) : tt;
    v8f acc[2][3];
#pragma unroll
    for (int u = 0; u < 2; ++u)
#pragma unroll
      for (int g = 0; g < 3; ++g) acc[u][g] = zero;

    // One LDS A-fragment per K-chunk feeds all 6 WMMAs (2 tiles x 3 gates).
    for (int kc = 0; kc < HQ; kc += 32) {
      v16h a = load16(aPtr + kc);
#pragma unroll
      for (int u = 0; u < 2; ++u)
#pragma unroll
        for (int g = 0; g < 3; ++g)
          acc[u][g] = wmma_f16(a, load16(bPtr[u][g] + kc), acc[u][g]);
    }
    __syncthreads();  // all waves finished reading old h (f16) for WMMA

#pragma unroll
    for (int u = 0; u < 2; ++u) {
      const float* xp = xg + (size_t)t * G3 + col[u];
#pragma unroll
      for (int r = 0; r < 8; ++r) {
        int b = r + mb;  // batch index (D layout: lane<16 -> M=r, lane>=16 -> M=r+8)
        const float* xq = xp + (size_t)b * TQ * G3;
        float xr = xq[0], xz = xq[HQ], xn = xq[2 * HQ];
        float hr = acc[u][0][r] + bias[u][0];
        float hz = acc[u][1][r] + bias[u][1];
        float hn = acc[u][2][r] + bias[u][2];
        float rg = 1.f / (1.f + __expf(-(xr + hr)));
        float zg = 1.f / (1.f + __expf(-(xz + hz)));
        float ng = tanhf(xn + rg * hn);
        float hold = hf[b * HQ + col[u]];   // only this wave touches these cols
        float hnew = (1.f - zg) * ng + zg * hold;
        hf[b * HQ + col[u]] = hnew;
        hh[b * HQ + col[u]] = (_Float16)hnew;
        outh[((size_t)b * TQ + t) * out_stride + out_off + col[u]] = (_Float16)hnew;
        if (outf) outf[((size_t)b * TQ + t) * HQ + col[u]] = hnew;
      }
    }
    __syncthreads();  // publish new h before next step
  }
}

// Bidirectional layer: block 0 = forward scan, block 1 = backward scan.
// The two scans are independent (separate xg, whh, h0; disjoint output halves),
// so they run concurrently on two WGPs, halving the serial critical path.
__global__ __launch_bounds__(512, 1) void gru_scan2(
    const _Float16* __restrict__ whhF, const float* __restrict__ bhhF,
    const float* __restrict__ xgF, const float* __restrict__ h0F,
    const _Float16* __restrict__ whhB, const float* __restrict__ bhhB,
    const float* __restrict__ xgB, const float* __restrict__ h0B,
    _Float16* __restrict__ outh) {
  __shared__ float    hf[BQ * HQ];
  __shared__ _Float16 hh[BQ * HQ];
  if (blockIdx.x == 0)
    gru_scan_core(hf, hh, whhF, bhhF, xgF, h0F, outh, 2 * HQ, 0, (float*)nullptr, 0);
  else
    gru_scan_core(hf, hh, whhB, bhhB, xgB, h0B, outh, 2 * HQ, HQ, (float*)nullptr, 1);
}

// Final unidirectional GRU: fp32 sequence output straight into d_out.
__global__ __launch_bounds__(512, 1) void gru_scan1(
    const _Float16* __restrict__ whh, const float* __restrict__ bhh,
    const float* __restrict__ xg, const float* __restrict__ h0,
    _Float16* __restrict__ outh, float* __restrict__ outf) {
  __shared__ float    hf[BQ * HQ];
  __shared__ _Float16 hh[BQ * HQ];
  gru_scan_core(hf, hh, whh, bhh, xg, h0, outh, 2 * HQ, 0, outf, 0);
}

__global__ void fc_head(const float* __restrict__ out2,  // [16,512,512]
                        const float* __restrict__ w,     // [1,512]
                        const float* __restrict__ bsc,   // [1]
                        float* __restrict__ out) {       // [16]
  int b = threadIdx.x;
  if (b < BQ) {
    float s = 0.f;
    const float* row = out2 + ((size_t)b * TQ + (TQ - 1)) * HQ;
    for (int h = 0; h < HQ; ++h) s += row[h] * w[h];
    s += bsc[0];
    out[b] = 1.f / (1.f + __expf(-s));
  }
}

extern "C" void kernel_launch(void* const* d_in, const int* in_sizes, int n_in,
                              void* d_out, int out_size, void* d_ws, size_t ws_size,
                              hipStream_t stream) {
  (void)in_sizes; (void)n_in; (void)out_size; (void)ws_size;
  const float* x  = (const float*)d_in[0];
  const float* h1 = (const float*)d_in[1];
  const float* h2 = (const float*)d_in[2];
  const int wihIdx[7] = {3, 7, 11, 15, 19, 23, 27};          // l0f,l0b,l1f,l1b,l2f,l2b,g2
  const int kin[7]    = {2048, 2048, 1024, 1024, 1024, 1024, 1024};

  char* ws = (char*)d_ws;
  size_t off = 0;
  auto allocH = [&](size_t n) { _Float16* p = (_Float16*)(ws + off); off += n * 2; return p; };
  auto allocF = [&](size_t n) { float*    p = (float*)(ws + off);    off += n * 4; return p; };

  _Float16* x16 = allocH((size_t)BQ * TQ * 2048);
  _Float16* wih16[7]; _Float16* whh16[7];
  for (int u = 0; u < 7; ++u) {
    wih16[u] = allocH((size_t)G3 * kin[u]);
    whh16[u] = allocH((size_t)G3 * HQ);
  }
  float*    xgF  = allocF((size_t)BQ * TQ * G3);
  float*    xgB  = allocF((size_t)BQ * TQ * G3);
  _Float16* bufA = allocH((size_t)BQ * TQ * 2 * HQ);
  _Float16* bufB = allocH((size_t)BQ * TQ * 2 * HQ);

  auto cvt = [&](const float* src, _Float16* dst, size_t n) {
    cvt_f32_f16<<<(int)((n + 255) / 256), 256, 0, stream>>>(src, dst, (int)n);
  };
  cvt(x, x16, (size_t)BQ * TQ * 2048);
  for (int u = 0; u < 7; ++u) {
    cvt((const float*)d_in[wihIdx[u]],     wih16[u], (size_t)G3 * kin[u]);
    cvt((const float*)d_in[wihIdx[u] + 1], whh16[u], (size_t)G3 * HQ);
  }

  const int GEMM_BLOCKS = (8192 / 16) * (1536 / 64) / 8;  // 1536 blocks of 8 waves
  float* out2 = (float*)d_out + BQ;                       // tuple: [logits(16) | out2(B,T,H)]

  const _Float16* Xin = x16;
  int K = 2048;
  _Float16* Yout = bufA;
  for (int l = 0; l < 3; ++l) {
    int uf = l * 2, ub = l * 2 + 1;
    gemm_xg<<<GEMM_BLOCKS, 256, 0, stream>>>(Xin, wih16[uf],
        (const float*)d_in[wihIdx[uf] + 2], xgF, K);
    gemm_xg<<<GEMM_BLOCKS, 256, 0, stream>>>(Xin, wih16[ub],
        (const float*)d_in[wihIdx[ub] + 2], xgB, K);
    gru_scan2<<<2, 512, 0, stream>>>(
        whh16[uf], (const float*)d_in[wihIdx[uf] + 3], xgF, h1 + (size_t)uf * BQ * HQ,
        whh16[ub], (const float*)d_in[wihIdx[ub] + 3], xgB, h1 + (size_t)ub * BQ * HQ,
        Yout);
    Xin = Yout; K = 2 * HQ;
    Yout = (Yout == bufA) ? bufB : bufA;
  }
  gemm_xg<<<GEMM_BLOCKS, 256, 0, stream>>>(Xin, wih16[6], (const float*)d_in[29], xgF, 1024);
  gru_scan1<<<1, 512, 0, stream>>>(whh16[6], (const float*)d_in[30], xgF, h2, Yout, out2);
  fc_head<<<1, 32, 0, stream>>>(out2, (const float*)d_in[31], (const float*)d_in[32],
                                (float*)d_out);
}
